// BahadanauAttention_68238440399012
// MI455X (gfx1250) — matrix operand order repaired and checked by reading, not compiled, not run on hardware
//
#include <hip/hip_runtime.h>
#include <hip/hip_bf16.h>

#define B_ 32
#define T_ 2048
#define D_ 512
#define Q_ 512

typedef __attribute__((ext_vector_type(16))) __bf16 v16bf;
typedef __attribute__((ext_vector_type(4)))  __bf16 v4bf;
typedef __attribute__((ext_vector_type(8)))  float  v8f;

__device__ inline v16bf pack16(float4 f0, float4 f1, float4 f2, float4 f3) {
  v16bf r;
  r[0]  = (__bf16)f0.x; r[1]  = (__bf16)f0.y; r[2]  = (__bf16)f0.z; r[3]  = (__bf16)f0.w;
  r[4]  = (__bf16)f1.x; r[5]  = (__bf16)f1.y; r[6]  = (__bf16)f1.z; r[7]  = (__bf16)f1.w;
  r[8]  = (__bf16)f2.x; r[9]  = (__bf16)f2.y; r[10] = (__bf16)f2.z; r[11] = (__bf16)f2.w;
  r[12] = (__bf16)f3.x; r[13] = (__bf16)f3.y; r[14] = (__bf16)f3.z; r[15] = (__bf16)f3.w;
  return r;
}

// ---------------------------------------------------------------------------
// Kernel 0: pre-convert w (f32, [D,D]) -> bf16 once. 0.5 MB, stays in L2.
// ---------------------------------------------------------------------------
__global__ __launch_bounds__(256)
void wcvt_kernel(const float* __restrict__ w, __bf16* __restrict__ wbf) {
  const int i = (blockIdx.x * 256 + threadIdx.x) * 4;
  float4 f = *(const float4*)(w + i);
  v4bf o;
  o[0] = (__bf16)f.x; o[1] = (__bf16)f.y; o[2] = (__bf16)f.z; o[3] = (__bf16)f.w;
  *(v4bf*)(wbf + i) = o;
}

// ---------------------------------------------------------------------------
// Kernel 1: qproj[b,e] = sum_q query[b,q] * u[e,q]   (B x D, tiny)
// ---------------------------------------------------------------------------
__global__ __launch_bounds__(D_)
void qproj_kernel(const float* __restrict__ query,
                  const float* __restrict__ u,
                  float* __restrict__ qproj) {
  const int b = blockIdx.x;
  const int e = threadIdx.x;
  const float* qrow = query + (size_t)b * Q_;
  const float* urow = u + (size_t)e * Q_;
  float acc = 0.f;
#pragma unroll 4
  for (int q = 0; q < Q_; ++q) acc = fmaf(qrow[q], urow[q], acc);
  qproj[(size_t)b * D_ + e] = acc;
}

// ---------------------------------------------------------------------------
// Kernel 2: masked score[b,t] = sum_e tanh( (out @ w^T)[b,t,e] + qproj[b,e] ) * v[e]
// WMMA bf16 GEMM with fused tanh + v-dot epilogue.
// Block = 4 waves x 16 rows = 64 rows of one batch. Wave holds its A tile
// (16 rows x 512 K, bf16) in registers; loops over 32 e-tiles of 16 columns.
// B comes pre-converted (bf16) so the inner loop is pure load+WMMA.
// ---------------------------------------------------------------------------
__global__ __launch_bounds__(128)
void score_kernel(const float* __restrict__ outp,
                  const __bf16* __restrict__ wbf,
                  const float* __restrict__ v,
                  const float* __restrict__ qproj,
                  const int* __restrict__ lengths,
                  float* __restrict__ score) {
  const int b    = blockIdx.y;
  const int wave = threadIdx.x >> 5;
  const int lane = threadIdx.x & 31;
  const int half = lane >> 4;   // 0: lanes 0-15, 1: lanes 16-31
  const int n    = lane & 15;   // A row within tile / B column within tile
  const int t0   = blockIdx.x * 64 + wave * 16;

  // ---- Load A tile (row t0+n, all 512 K) into registers as bf16 ----
  // A layout (16-bit 16x32): half 0 elems 0..15 -> K=0..15
  //                          half 1 elems 0..15 -> K=16..31
  const float* arow = outp + ((size_t)b * T_ + (size_t)(t0 + n)) * D_;
  v16bf A[16];
#pragma unroll
  for (int kc = 0; kc < 16; ++kc) {
    const float* p = arow + kc * 32 + half * 16;
    float4 f0 = *(const float4*)(p);
    float4 f1 = *(const float4*)(p + 4);
    float4 f2 = *(const float4*)(p + 8);
    float4 f3 = *(const float4*)(p + 12);
    A[kc] = pack16(f0, f1, f2, f3);
  }

  float sacc[8] = {0.f, 0.f, 0.f, 0.f, 0.f, 0.f, 0.f, 0.f};

  for (int et = 0; et < 32; ++et) {
    // B tile: column e = et*16 + n is row (et*16+n) of w (bf16).
    // B k order used here (same as A, not the ISA table order): lane elems i=0..15 -> K = half*16 + kc*32 + i.
    const __bf16* brow = wbf + ((size_t)(et * 16 + n)) * D_ + half * 16;
    // prefetch next e-tile's w row (wraps; harmless)
    __builtin_prefetch(wbf + ((size_t)((((et + 1) & 31) * 16) + n)) * D_, 0, 3);

    v8f c = {};
#pragma unroll
    for (int kc = 0; kc < 16; ++kc) {
      v16bf Bv = *(const v16bf*)(brow + kc * 32);  // 32B aligned, 2x b128
      c = __builtin_amdgcn_wmma_f32_16x16x32_bf16(
          /*neg_a=*/false, A[kc], /*neg_b=*/false, Bv,
          /*c_mod=*/(short)0, c, /*reuse_a=*/false, /*reuse_b=*/false);
    }

    // Epilogue: C layout -> this lane holds column N = n, rows M = half*8 + r.
    const int e  = et * 16 + n;
    const float qp = qproj[(size_t)b * D_ + e];
    const float vv = v[e];
#pragma unroll
    for (int r = 0; r < 8; ++r)
      sacc[r] += tanhf(c[r] + qp) * vv;
  }

  // Reduce over the 16 columns held by each 16-lane half.
#pragma unroll
  for (int r = 0; r < 8; ++r) {
    float s = sacc[r];
    s += __shfl_xor(s, 8, 16);
    s += __shfl_xor(s, 4, 16);
    s += __shfl_xor(s, 2, 16);
    s += __shfl_xor(s, 1, 16);
    sacc[r] = s;
  }

  if (n == 0) {
    const int len = lengths[b];
#pragma unroll
    for (int r = 0; r < 8; ++r) {
      const int t = t0 + half * 8 + r;
      score[(size_t)b * T_ + t] = (t < len) ? sacc[r] : -__builtin_inff();
    }
  }
}

// ---------------------------------------------------------------------------
// Kernel 3: weights[b,:] = softmax(score[b,:]) over T (exp(-inf) -> 0)
// ---------------------------------------------------------------------------
__global__ __launch_bounds__(256)
void softmax_kernel(const float* __restrict__ score,
                    float* __restrict__ weights) {
  const int b = blockIdx.x, tid = threadIdx.x;
  __shared__ float red[256];
  const float* s = score + (size_t)b * T_;

  float m = -__builtin_inff();
  for (int t = tid; t < T_; t += 256) m = fmaxf(m, s[t]);
  red[tid] = m; __syncthreads();
  for (int off = 128; off > 0; off >>= 1) {
    if (tid < off) red[tid] = fmaxf(red[tid], red[tid + off]);
    __syncthreads();
  }
  const float smax = red[0];
  __syncthreads();

  float sum = 0.f;
  for (int t = tid; t < T_; t += 256) sum += __expf(s[t] - smax);
  red[tid] = sum; __syncthreads();
  for (int off = 128; off > 0; off >>= 1) {
    if (tid < off) red[tid] += red[tid + off];
    __syncthreads();
  }
  const float inv = 1.0f / red[0];

  float* wr = weights + (size_t)b * T_;
  for (int t = tid; t < T_; t += 256) wr[t] = __expf(s[t] - smax) * inv;
}

// ---------------------------------------------------------------------------
// Kernel 4a: partial expectation over a T-slice of 256 rows.
// grid (8, B), block 512 (thread = d). partial[(b*8+s), d] -> ws.
// ---------------------------------------------------------------------------
__global__ __launch_bounds__(D_)
void expect_part_kernel(const float* __restrict__ outp,
                        const float* __restrict__ weights,
                        float* __restrict__ partial) {
  const int s = blockIdx.x, b = blockIdx.y, d = threadIdx.x;
  const float* wr   = weights + (size_t)b * T_ + s * 256;
  const float* orow = outp + ((size_t)b * T_ + (size_t)s * 256) * D_ + d;
  float acc = 0.f;
#pragma unroll 4
  for (int t = 0; t < 256; ++t) acc = fmaf(wr[t], orow[(size_t)t * D_], acc);
  partial[((size_t)b * 8 + s) * D_ + d] = acc;
}

// ---------------------------------------------------------------------------
// Kernel 4b: deterministic reduce of the 8 slices.
// ---------------------------------------------------------------------------
__global__ __launch_bounds__(D_)
void expect_reduce_kernel(const float* __restrict__ partial,
                          float* __restrict__ eout) {
  const int b = blockIdx.x, d = threadIdx.x;
  float acc = 0.f;
#pragma unroll
  for (int s = 0; s < 8; ++s) acc += partial[((size_t)b * 8 + s) * D_ + d];
  eout[(size_t)b * D_ + d] = acc;
}

extern "C" void kernel_launch(void* const* d_in, const int* in_sizes, int n_in,
                              void* d_out, int out_size, void* d_ws, size_t ws_size,
                              hipStream_t stream) {
  (void)in_sizes; (void)n_in; (void)out_size; (void)ws_size;
  const float* outp    = (const float*)d_in[0];  // [B,T,D]
  const float* query   = (const float*)d_in[1];  // [B,Q]
  const int*   lengths = (const int*)  d_in[2];  // [B]
  const float* w       = (const float*)d_in[3];  // [D,D]
  const float* u       = (const float*)d_in[4];  // [D,Q]
  const float* v       = (const float*)d_in[5];  // [D,1]

  float* score   = (float*)d_out;                    // [B,T]
  float* weights = score + (size_t)B_ * T_;          // [B,T]
  float* expect  = weights + (size_t)B_ * T_;        // [B,D]

  // workspace layout
  float*  qproj   = (float*)d_ws;                                     // 64 KB
  __bf16* wbf     = (__bf16*)((char*)d_ws + (size_t)B_ * D_ * 4);     // 512 KB
  float*  partial = (float*)((char*)wbf + (size_t)D_ * D_ * 2);       // 512 KB

  wcvt_kernel<<<(D_ * D_) / (256 * 4), 256, 0, stream>>>(w, wbf);
  qproj_kernel<<<B_, D_, 0, stream>>>(query, u, qproj);
  score_kernel<<<dim3(T_ / 64, B_), 128, 0, stream>>>(outp, wbf, v, qproj, lengths, score);
  softmax_kernel<<<B_, 256, 0, stream>>>(score, weights);
  expect_part_kernel<<<dim3(8, B_), D_, 0, stream>>>(outp, weights, partial);
  expect_reduce_kernel<<<B_, D_, 0, stream>>>(partial, expect);
}
